// SelfAttention_326417515366
// MI455X (gfx1250) — matrix operand order repaired and checked by reading, not compiled, not run on hardware
//
#include <hip/hip_runtime.h>
#include <hip/hip_bf16.h>

// ---------------------------------------------------------------------------
// CDNA5 (gfx1250) bf16-WMMA self-attention pipeline.
//   k0: fp32 -> bf16 weight conversion
//   k1: per-head projection  Y = X @ W^T   (Q,K row-major; V transposed)
//   k2: flash attention (online softmax), 128-key blocks, bf16 WMMA, fp32 accum
//   k3: output projection  out = AO @ Wo^T + bo
// ---------------------------------------------------------------------------

typedef __bf16 v16bf __attribute__((ext_vector_type(16)));
typedef __bf16 v8bf  __attribute__((ext_vector_type(8)));
typedef float  v8f   __attribute__((ext_vector_type(8)));

#define WMMA_BF16(A, B, C) \
  __builtin_amdgcn_wmma_f32_16x16x32_bf16(false, (A), false, (B), (short)0, (C), false, false)

// A-matrix 16x32 bf16 fragment: lane = M row (lane&15);
// lanes 0-15 hold K = kb..kb+15, lanes 16-31 hold K = kb+16..kb+31.
static __device__ __forceinline__ v16bf ldA_bf16(const __bf16* base, int stride,
                                                 int lane, int kbase) {
  const __bf16* p = base + (size_t)(lane & 15) * stride + kbase + ((lane & 16) ? 16 : 0);
  v8bf lo = *(const v8bf*)p;
  v8bf hi = *(const v8bf*)(p + 8);
  v16bf r;
#pragma unroll
  for (int i = 0; i < 8; ++i) { r[i] = lo[i]; r[i + 8] = hi[i]; }
  return r;
}

// Same A fragment but sourced from fp32 (convert inline).
static __device__ __forceinline__ v16bf ldA_f32(const float* base, int stride,
                                                int lane, int kbase) {
  const float* p = base + (size_t)(lane & 15) * stride + kbase + ((lane & 16) ? 16 : 0);
  v16bf r;
#pragma unroll
  for (int i = 0; i < 8; ++i) { r[i] = (__bf16)p[i]; r[i + 8] = (__bf16)p[i + 8]; }
  return r;
}

// B-matrix 32x16 bf16 fragment, stored as B^T row-major [N][K]:
// lane = N col (lane&15); lanes 0-15 hold K 0-15, lanes 16-31 hold K 16-31.
static __device__ __forceinline__ v16bf ldB_bf16(const __bf16* base, int stride,
                                                 int lane, int kbase) {
  const __bf16* p = base + (size_t)(lane & 15) * stride + kbase + ((lane & 16) ? 16 : 0);
  return *(const v16bf*)p;  // 32B -> 2x global_load_b128
}

// ---------------------------------------------------------------------------
__global__ void cvt_kernel(const float* __restrict__ src, __bf16* __restrict__ dst, int n) {
  int i = blockIdx.x * blockDim.x + threadIdx.x;
  if (i < n) dst[i] = (__bf16)src[i];
}

// ---------------------------------------------------------------------------
// Projection: Y[n,h,s,:] = X[n,s,h*64:+64] @ W^T  (64x64 per head, K=64).
// Block = 256 thr (8 waves), covers 64 s-rows x 64 d-cols; wave does 2 tiles.
// transposed==0 -> dst[nh][s][d] ; transposed==1 -> dst[nh][d][s] (for V).
__global__ __launch_bounds__(256) void proj_kernel(const float* __restrict__ X,
                                                   const __bf16* __restrict__ Wb,
                                                   __bf16* __restrict__ dst,
                                                   int transposed) {
  const int lane = threadIdx.x & 31;
  const int wave = threadIdx.x >> 5;
  const int nh = blockIdx.y;
  const int n = nh >> 4, h = nh & 15;
  const int s0 = blockIdx.x * 64;
  const int lrow = (lane & 16) ? 8 : 0, lcol = lane & 15;
#pragma unroll
  for (int tt = 0; tt < 2; ++tt) {
    const int t = wave * 2 + tt, tm = t >> 2, tn = t & 3;
    const float* Xb = X + ((size_t)(n * 1024 + s0 + tm * 16)) * 1024 + h * 64;
    v16bf a0 = ldA_f32(Xb, 1024, lane, 0);
    v16bf a1 = ldA_f32(Xb, 1024, lane, 32);
    const __bf16* Bb = Wb + tn * 16 * 64;  // W row-major == B^T rows
    v8f c = {};
    c = WMMA_BF16(a0, ldB_bf16(Bb, 64, lane, 0), c);
    c = WMMA_BF16(a1, ldB_bf16(Bb, 64, lane, 32), c);
    if (!transposed) {
      __bf16* d = dst + ((size_t)nh * 1024 + s0 + tm * 16 + lrow) * 64 + tn * 16 + lcol;
#pragma unroll
      for (int r = 0; r < 8; ++r) d[(size_t)r * 64] = (__bf16)c[r];
    } else {
      v8bf pk;
#pragma unroll
      for (int r = 0; r < 8; ++r) pk[r] = (__bf16)c[r];
      __bf16* d = dst + ((size_t)nh * 64 + tn * 16 + lcol) * 1024 + s0 + tm * 16 + lrow;
      *(v8bf*)d = pk;  // contiguous along s -> one b128 store
    }
  }
}

// ---------------------------------------------------------------------------
// Flash attention per (n,h). Block = 8 waves x 16 query rows = 128 queries.
// Each wave streams causal key blocks of 128 (32 WMMA per block) and keeps
// fp32 online-softmax stats; P is transposed through LDS for the PV matmul.
__global__ __launch_bounds__(256) void attn_kernel(const __bf16* __restrict__ Qp,
                                                   const __bf16* __restrict__ Kp,
                                                   const __bf16* __restrict__ Vt,
                                                   __bf16* __restrict__ AO) {
  __shared__ __bf16 Pl[8][16][128];
  const int lane = threadIdx.x & 31;
  const int wave = threadIdx.x >> 5;
  const int nh = blockIdx.y;
  const int qwbase = blockIdx.x * 128 + wave * 16;
  const int lrow = (lane & 16) ? 8 : 0, lcol = lane & 15;

  const __bf16* Qb = Qp + ((size_t)nh * 1024 + qwbase) * 64;
  const v16bf aq0 = ldA_bf16(Qb, 64, lane, 0);
  const v16bf aq1 = ldA_bf16(Qb, 64, lane, 32);
  const __bf16* KbBase = Kp + (size_t)nh * 1024 * 64;
  const __bf16* Vb = Vt + (size_t)nh * 64 * 1024;
  __bf16* Pw = &Pl[wave][0][0];

  v8f acc0 = {}, acc1 = {}, acc2 = {}, acc3 = {};
  float m2[8], l[8];
#pragma unroll
  for (int r = 0; r < 8; ++r) { m2[r] = -1e30f; l[r] = 0.f; }

  const float SC = 0.04508422002778011f;  // log2(e) / sqrt(EMBED=1024)
  const int kb_last = qwbase >> 7;
  for (int kb = 0; kb <= kb_last; ++kb) {
    const int k0 = kb * 128;
    // ---- scores: 16q x 128k, K-dim = 64 -> 16 WMMA ----
    v8f c[8] = {};
#pragma unroll
    for (int t = 0; t < 8; ++t) {
      const __bf16* Kb = KbBase + (size_t)(k0 + t * 16) * 64;
      c[t] = WMMA_BF16(aq0, ldB_bf16(Kb, 64, lane, 0), c[t]);
      c[t] = WMMA_BF16(aq1, ldB_bf16(Kb, 64, lane, 32), c[t]);
    }
    // ---- scale + causal mask (diagonal block only) ----
    const bool diag = (kb == kb_last);
#pragma unroll
    for (int t = 0; t < 8; ++t) {
      const int key = k0 + t * 16 + lcol;
#pragma unroll
      for (int r = 0; r < 8; ++r) {
        float v = c[t][r] * SC;
        if (diag && key > qwbase + lrow + r) v = -1e30f;
        c[t][r] = v;
      }
    }
    // ---- row max (butterfly over the 16-lane half) + rescale factors ----
    float alpha[8];
#pragma unroll
    for (int r = 0; r < 8; ++r) {
      float m = fmaxf(fmaxf(fmaxf(c[0][r], c[1][r]), fmaxf(c[2][r], c[3][r])),
                      fmaxf(fmaxf(c[4][r], c[5][r]), fmaxf(c[6][r], c[7][r])));
      m = fmaxf(m, __shfl_xor(m, 1, 32));
      m = fmaxf(m, __shfl_xor(m, 2, 32));
      m = fmaxf(m, __shfl_xor(m, 4, 32));
      m = fmaxf(m, __shfl_xor(m, 8, 32));
      const float mn = fmaxf(m2[r], m);
      alpha[r] = exp2f(m2[r] - mn);
      m2[r] = mn;
    }
    // ---- P = exp2(s - m); row sums; stash P (bf16) in LDS for transpose ----
    float rs[8];
#pragma unroll
    for (int r = 0; r < 8; ++r) rs[r] = 0.f;
#pragma unroll
    for (int t = 0; t < 8; ++t) {
#pragma unroll
      for (int r = 0; r < 8; ++r) {
        const float p = exp2f(c[t][r] - m2[r]);
        rs[r] += p;
        Pw[(lrow + r) * 128 + t * 16 + lcol] = (__bf16)p;
      }
    }
#pragma unroll
    for (int r = 0; r < 8; ++r) {
      float s = rs[r];
      s += __shfl_xor(s, 1, 32);
      s += __shfl_xor(s, 2, 32);
      s += __shfl_xor(s, 4, 32);
      s += __shfl_xor(s, 8, 32);
      l[r] = l[r] * alpha[r] + s;
      acc0[r] *= alpha[r]; acc1[r] *= alpha[r];
      acc2[r] *= alpha[r]; acc3[r] *= alpha[r];
    }
    // ---- P (C-layout) -> A fragments via LDS; same-wave RAW needs dscnt ----
    asm volatile("s_wait_dscnt 0" ::: "memory");
    v16bf ap[4];
#pragma unroll
    for (int kk = 0; kk < 4; ++kk) ap[kk] = ldA_bf16(Pw, 128, lane, kk * 32);
    // ---- PV: Vt rows [d][s] are B^T rows; K-dim = 128 keys -> 16 WMMA ----
#pragma unroll
    for (int kk = 0; kk < 4; ++kk) {
      acc0 = WMMA_BF16(ap[kk], ldB_bf16(Vb, 1024, lane, k0 + kk * 32), acc0);
      acc1 = WMMA_BF16(ap[kk], ldB_bf16(Vb + 16 * 1024, 1024, lane, k0 + kk * 32), acc1);
      acc2 = WMMA_BF16(ap[kk], ldB_bf16(Vb + 32 * 1024, 1024, lane, k0 + kk * 32), acc2);
      acc3 = WMMA_BF16(ap[kk], ldB_bf16(Vb + 48 * 1024, 1024, lane, k0 + kk * 32), acc3);
    }
  }
  // ---- normalize and store AO[n][s][h*64+d] as bf16 ----
  const int n = nh >> 4, h = nh & 15;
#pragma unroll
  for (int r = 0; r < 8; ++r) {
    const float inv = 1.f / l[r];
    const size_t row = (size_t)(n * 1024 + qwbase + lrow + r) * 1024 + h * 64 + lcol;
    AO[row]      = (__bf16)(acc0[r] * inv);
    AO[row + 16] = (__bf16)(acc1[r] * inv);
    AO[row + 32] = (__bf16)(acc2[r] * inv);
    AO[row + 48] = (__bf16)(acc3[r] * inv);
  }
}

// ---------------------------------------------------------------------------
// Output projection: out[4096,1024] = AO @ Wo^T + bo (fp32 out).
// Block = 8 waves; wave = 16 rows x 64 cols; K loop over 1024.
__global__ __launch_bounds__(256) void outproj_kernel(const __bf16* __restrict__ AO,
                                                      const __bf16* __restrict__ Wob,
                                                      const float* __restrict__ bo,
                                                      float* __restrict__ out) {
  const int lane = threadIdx.x & 31;
  const int wave = threadIdx.x >> 5;
  const int rb = blockIdx.y * 128 + wave * 16;
  const int cb = blockIdx.x * 64;
  v8f acc[4] = {};
  const __bf16* Ab = AO + (size_t)rb * 1024;
  for (int k = 0; k < 1024; k += 32) {
    const v16bf a = ldA_bf16(Ab, 1024, lane, k);
#pragma unroll
    for (int t = 0; t < 4; ++t) {
      const v16bf b = ldB_bf16(Wob + (size_t)(cb + t * 16) * 1024, 1024, lane, k);
      acc[t] = WMMA_BF16(a, b, acc[t]);
    }
  }
  const int lrow = (lane & 16) ? 8 : 0, lcol = lane & 15;
#pragma unroll
  for (int t = 0; t < 4; ++t) {
    const int col = cb + t * 16 + lcol;
    const float bias = bo[col];
#pragma unroll
    for (int r = 0; r < 8; ++r)
      out[(size_t)(rb + lrow + r) * 1024 + col] = acc[t][r] + bias;
  }
}

// ---------------------------------------------------------------------------
extern "C" void kernel_launch(void* const* d_in, const int* in_sizes, int n_in,
                              void* d_out, int out_size, void* d_ws, size_t ws_size,
                              hipStream_t stream) {
  (void)in_sizes; (void)n_in; (void)out_size; (void)ws_size;
  const float* values = (const float*)d_in[0];
  const float* keys   = (const float*)d_in[1];
  const float* query  = (const float*)d_in[2];
  /* d_in[3] = mask: causal by construction, applied analytically */
  const float* Wv = (const float*)d_in[4];
  const float* Wk = (const float*)d_in[5];
  const float* Wq = (const float*)d_in[6];
  const float* Wo = (const float*)d_in[7];
  const float* bo = (const float*)d_in[8];

  char* base = (char*)d_ws;
  size_t off = 0;
  auto take = [&](size_t bytes) -> char* {
    char* p = base + off;
    off = (off + bytes + 255) & ~(size_t)255;
    return p;
  };
  __bf16* Wvb = (__bf16*)take(64 * 64 * 2);
  __bf16* Wkb = (__bf16*)take(64 * 64 * 2);
  __bf16* Wqb = (__bf16*)take(64 * 64 * 2);
  __bf16* Wob = (__bf16*)take((size_t)1024 * 1024 * 2);
  const size_t qkv_bytes = (size_t)4 * 16 * 1024 * 64 * 2;  // [n][h][s][d] bf16
  __bf16* Qp = (__bf16*)take(qkv_bytes);
  __bf16* Kp = (__bf16*)take(qkv_bytes);
  __bf16* Vt = (__bf16*)take(qkv_bytes);                    // [n][h][d][s]
  __bf16* AO = (__bf16*)take((size_t)4 * 1024 * 1024 * 2);  // [n*s][e]

  cvt_kernel<<<16, 256, 0, stream>>>(Wv, Wvb, 64 * 64);
  cvt_kernel<<<16, 256, 0, stream>>>(Wk, Wkb, 64 * 64);
  cvt_kernel<<<16, 256, 0, stream>>>(Wq, Wqb, 64 * 64);
  cvt_kernel<<<4096, 256, 0, stream>>>(Wo, Wob, 1024 * 1024);

  proj_kernel<<<dim3(16, 64), 256, 0, stream>>>(query, Wqb, Qp, 0);
  proj_kernel<<<dim3(16, 64), 256, 0, stream>>>(keys, Wkb, Kp, 0);
  proj_kernel<<<dim3(16, 64), 256, 0, stream>>>(values, Wvb, Vt, 1);

  attn_kernel<<<dim3(8, 64), 256, 0, stream>>>(Qp, Kp, Vt, AO);

  outproj_kernel<<<dim3(16, 32), 256, 0, stream>>>(AO, Wob, bo, (float*)d_out);
}
